// SS2D_Mamba_55190329754120
// MI455X (gfx1250) — matrix operand order repaired and checked by reading, not compiled, not run on hardware
//
#include <hip/hip_runtime.h>
#include <hip/hip_bf16.h>

typedef __attribute__((ext_vector_type(16))) __bf16        v16bf;
typedef __attribute__((ext_vector_type(2)))  __bf16        bf16x2;
typedef __attribute__((ext_vector_type(8)))  float         v8f;
typedef __attribute__((ext_vector_type(4)))  unsigned int  u32x4;

#define Bn 8
#define Cc 256
#define Hh 96
#define Ww 96
#define OUT4 (4 * Cc)             // 1024
#define XS_PITCH 264              // 256 + 8 bf16 elements -> 528B row pitch (bank spread)
#define PLANE (Ww * XS_PITCH)     // 25344 bf16 elements
// LDS layout (bf16 elements):
//   [0,        PLANE)   : xs tile  [w][c]
//   [PLANE,  3*PLANE)   : interleaved (A,B) gate pairs: idx = PLANE + 2*(w*PITCH+c) + {0,1}
//   [3*PLANE,5*PLANE)   : interleaved (C,D) gate pairs
#define LDS_BYTES (PLANE * 5 * 2) // 253440 B (< 320KB/WGP)

union Frag { u32x4 u[2]; v16bf v; };

// ---- pre-pass: weight fp32 [1024,256] -> bf16 row-major in workspace ----
__global__ void ss2d_wcvt_kernel(const float* __restrict__ w,
                                 __bf16* __restrict__ wb, int n) {
    int i = blockIdx.x * blockDim.x + threadIdx.x;
    if (i < n) wb[i] = (__bf16)w[i];
}

// ---- fused: GEMM(params=sigmoid(xs*W^T+bias)) in LDS + selective scan ----
__global__ __launch_bounds__(256)
void ss2d_fused_kernel(const float* __restrict__ x,
                       const __bf16* __restrict__ wb,   // [1024,256] bf16
                       const float* __restrict__ bias,  // [1024] fp32
                       float* __restrict__ out) {
    extern __shared__ __bf16 lds[];
    __bf16* xs = lds;                          // [96][XS_PITCH] bf16

    const int n    = blockIdx.x;               // sequence id 0..767
    const int b    = n / Hh;
    const int hrow = n % Hh;
    const int tid  = threadIdx.x;              // 0..255
    const int lane = tid & 31;
    const int wave = tid >> 5;                 // 0..7

    // ---------- Phase 0: x[b, :, hrow, :] -> bf16 LDS tile [w][c] ----------
    {
        const float* xbase = x + ((size_t)b * Cc * Hh + (size_t)hrow) * Ww;
        for (int idx = tid; idx < Cc * (Ww / 4); idx += 256) {
            const int c  = idx / (Ww / 4);
            const int w4 = idx % (Ww / 4);
            const float4 f = *(const float4*)(xbase + (size_t)c * (Hh * Ww) + w4 * 4);
            const int w0 = w4 * 4;
            xs[(w0 + 0) * XS_PITCH + c] = (__bf16)f.x;
            xs[(w0 + 1) * XS_PITCH + c] = (__bf16)f.y;
            xs[(w0 + 2) * XS_PITCH + c] = (__bf16)f.z;
            xs[(w0 + 3) * XS_PITCH + c] = (__bf16)f.w;
        }
    }
    __syncthreads();

    // ---------- Phase 1: WMMA GEMM, sigmoid, scatter to gate pair-planes ---
    // wave owns n-tiles [wave*8, wave*8+8); m-tiles processed in pairs with two
    // independent accumulator chains to hide WMMA latency; K=256 in 8 steps.
    for (int nt = wave * 8; nt < wave * 8 + 8; ++nt) {
        const int o0 = nt * 16;                 // output column base (0..1008)
        const int oc = o0 + (lane & 15);        // this lane's column
        const float bv   = bias[oc];
        const int  gate  = o0 >> 8;             // whole tile in one gate (256%16==0)
        const int  cbase = o0 & 255;            // channel base within gate
        // interleaved pair-plane base for this gate (element units)
        __bf16* plane = lds + (size_t)PLANE * (1 + 2 * (gate >> 1)) + (gate & 1);

        // B fragments (weight) hoisted: reused across all 6 m-tiles, from L2
        Frag bf[8];
        {
            const __bf16* wrow = wb + (size_t)oc * Cc + ((lane >> 4) << 3);
            #pragma unroll
            for (int k = 0; k < 8; ++k) {
                bf[k].u[0] = *(const u32x4*)(wrow + k * 32);
                bf[k].u[1] = *(const u32x4*)(wrow + k * 32 + 16);
            }
        }
        for (int mt = 0; mt < 6; mt += 2) {
            const int m0 = mt * 16;
            v8f acc0 = {};
            v8f acc1 = {};
            #pragma unroll
            for (int k = 0; k < 8; ++k) {
                const int koff = k * 32 + ((lane >> 4) << 3);
                const __bf16* arow0 = xs + (size_t)(m0      + (lane & 15)) * XS_PITCH + koff;
                const __bf16* arow1 = xs + (size_t)(m0 + 16 + (lane & 15)) * XS_PITCH + koff;
                Frag a0, a1;
                a0.u[0] = *(const u32x4*)(arow0);
                a0.u[1] = *(const u32x4*)(arow0 + 16);
                a1.u[0] = *(const u32x4*)(arow1);
                a1.u[1] = *(const u32x4*)(arow1 + 16);
                acc0 = __builtin_amdgcn_wmma_f32_16x16x32_bf16(
                        false, a0.v, false, bf[k].v, (short)0, acc0, false, false);
                acc1 = __builtin_amdgcn_wmma_f32_16x16x32_bf16(
                        false, a1.v, false, bf[k].v, (short)0, acc1, false, false);
            }
            // C layout: lane l, vgpr v -> (M = m0 + v + 8*(l/16), N = o0 + l%16)
            const int mbase = m0 + ((lane >> 4) << 3);
            const int ccol  = cbase + (lane & 15);
            #pragma unroll
            for (int v = 0; v < 8; ++v) {
                const float s0 = acc0[v] + bv;
                const float s1 = acc1[v] + bv;
                // sigmoid via v_exp_f32 + v_rcp_f32 (bf16 output -> rcp is exact enough)
                const float p0 = __builtin_amdgcn_rcpf(1.0f + __expf(-s0));
                const float p1 = __builtin_amdgcn_rcpf(1.0f + __expf(-s1));
                plane[2 * ((mbase + v     ) * XS_PITCH + ccol)] = (__bf16)p0;
                plane[2 * ((mbase + v + 16) * XS_PITCH + ccol)] = (__bf16)p1;
            }
        }
    }
    __syncthreads();

    // ---------- Phase 2: selective scan along w, one channel per thread ---
    {
        const int c = tid;                       // channel 0..255
        const __bf16* pAB = lds + PLANE;         // (a,b) pairs
        const __bf16* pCD = lds + 3 * PLANE;     // (c,d) pairs
        float h = 0.0f;
        float* obase = out + ((size_t)(b * Cc + c) * Hh + hrow) * Ww;
        for (int w = 0; w < Ww; w += 4) {
            float yv[4];
            #pragma unroll
            for (int j = 0; j < 4; ++j) {
                const int off = (w + j) * XS_PITCH + c;
                const bf16x2 ab = *(const bf16x2*)(pAB + 2 * off); // ds_load_b32
                const bf16x2 cd = *(const bf16x2*)(pCD + 2 * off); // ds_load_b32
                const float xv = (float)xs[off];
                h = (float)ab.x * h + (float)ab.y * xv;
                yv[j] = (float)cd.x * h + (float)cd.y * xv;
            }
            *(float4*)(obase + w) = make_float4(yv[0], yv[1], yv[2], yv[3]);
        }
    }
}

extern "C" void kernel_launch(void* const* d_in, const int* in_sizes, int n_in,
                              void* d_out, int out_size, void* d_ws, size_t ws_size,
                              hipStream_t stream) {
    const float* x      = (const float*)d_in[0];   // [8,256,96,96]
    const float* weight = (const float*)d_in[1];   // [1024,256]
    const float* bias   = (const float*)d_in[2];   // [1024]
    float*       out    = (float*)d_out;           // [8,256,96,96]
    __bf16*      wbf    = (__bf16*)d_ws;           // 512KB bf16 weight, L2-resident

    const int wn = OUT4 * Cc;                      // 262144
    ss2d_wcvt_kernel<<<(wn + 255) / 256, 256, 0, stream>>>(weight, wbf, wn);

    const int nseq = Bn * Hh;                      // 768 workgroups, 1 sequence each
    ss2d_fused_kernel<<<nseq, 256, LDS_BYTES, stream>>>(x, wbf, bias, out);
}
